// SelfAttention_74672301408699
// MI455X (gfx1250) — matrix operand order repaired and checked by reading, not compiled, not run on hardware
//
#include <hip/hip_runtime.h>

typedef _Float16 v16h __attribute__((ext_vector_type(16)));
typedef _Float16 v8h  __attribute__((ext_vector_type(8)));
typedef _Float16 h2   __attribute__((ext_vector_type(2)));
typedef float    v8f  __attribute__((ext_vector_type(8)));

#define B_ 2
#define T_ 2048
#define H_ 16
#define D_ 128
#define SCALE_ 0.08838834764831845f
#define NEG_ -10000.0f

#define KSTRIDE 136   // halves per K row  (128 data + 8 pad, keeps 16B align)
#define VSTRIDE 40    // halves per V^T row (32 data + 8 pad)

static __device__ __forceinline__ v8f wmma16(v16h a, v16h b, v8f c) {
  // D = A(16x32 f16) * B(32x16 f16) + C(16x16 f32)
  return __builtin_amdgcn_wmma_f32_16x16x32_f16(false, a, false, b, (short)0, c, false, false);
}

__global__ __launch_bounds__(128) void fa_alibi_bias_kernel(
    const float* __restrict__ q, const float* __restrict__ k,
    const float* __restrict__ v, const int* __restrict__ cum,
    const float* __restrict__ slopes, const float* __restrict__ bias,
    float* __restrict__ out)
{
  __shared__ _Float16 sK[32 * KSTRIDE];       // K tile, f16, [key][d]
  __shared__ _Float16 sV[D_ * VSTRIDE];       // V tile, f16, transposed [d][key]
  __shared__ _Float16 sP[4 * 2 * 16 * 32];    // per-wave, per-Mtile P transpose [m][k]

  const int tid  = threadIdx.x;
  const int wave = tid >> 5;
  const int lane = tid & 31;
  const int n    = lane & 15;   // N index / A-row index
  const int hi   = lane >> 4;   // lane-half select
  const int hi8  = hi * 8;
  const int hi16 = hi * 16;

  const int tiles = T_ / 128;                 // 128 query rows per workgroup
  const int bh = blockIdx.x / tiles;
  const int mt = blockIdx.x % tiles;
  const int b  = bh / H_;
  const int h  = bh % H_;
  const int m_base = mt * 128 + wave * 32;    // this wave's 32 query rows (2 M-tiles)

  const int   tlen  = cum[b + 1] - cum[b];
  const float slope = slopes[h];

  // all-ones B matrix: row-sum(P) = P x Ones via one WMMA per M-tile
  v16h ones;
#pragma unroll
  for (int j = 0; j < 16; ++j) ones[j] = (_Float16)1.0f;

  // ---- Q tiles (2 x 16 x 128) -> WMMA A layout registers, pre-scaled by 1/sqrt(D) ----
  // A-layout: lane m = lane&15; element j -> K = (j>>3)*16 + hi*8 + (j&7)
  v16h qa[2][4];
#pragma unroll
  for (int mm = 0; mm < 2; ++mm) {
    const float* qrow =
        q + (((size_t)b * T_ + (size_t)(m_base + mm * 16 + n)) * H_ + h) * D_;
#pragma unroll
    for (int c = 0; c < 4; ++c) {
      float t0[8], t1[8];
      *(float4*)&t0[0] = *(const float4*)(qrow + 32 * c + hi8);
      *(float4*)&t0[4] = *(const float4*)(qrow + 32 * c + hi8 + 4);
      *(float4*)&t1[0] = *(const float4*)(qrow + 32 * c + 16 + hi8);
      *(float4*)&t1[4] = *(const float4*)(qrow + 32 * c + 16 + hi8 + 4);
#pragma unroll
      for (int j = 0; j < 8; ++j) {
        qa[mm][c][j]     = (_Float16)(t0[j] * SCALE_);
        qa[mm][c][j + 8] = (_Float16)(t1[j] * SCALE_);
      }
    }
  }

  v8f acc[2][8];                    // 2 x (16 rows x 128 cols) accumulators (C-layout)
  float mrun[2][8], lrun[2][8];     // per-row running max / sum (row = mm*16 + r + hi8)
#pragma unroll
  for (int mm = 0; mm < 2; ++mm)
#pragma unroll
    for (int r = 0; r < 8; ++r) {
      acc[mm][r] = 0.0f; mrun[mm][r] = -1e30f; lrun[mm][r] = 0.0f;
    }

  // staging role: 4 threads per key row, 32 floats each
  const int row_s = tid >> 2;
  const int part  = tid & 3;
  const int d0    = part * 32;
  const size_t kv_base = (((size_t)b * T_ + row_s) * H_ + h) * D_ + d0;

  for (int s = 0; s < T_ / 32; ++s) {
    const int key0 = s * 32;
    __syncthreads();   // previous tile fully consumed
    {
      const float* kp = k + kv_base + (size_t)key0 * H_ * D_;
      const float* vp = v + kv_base + (size_t)key0 * H_ * D_;
      float kt[32], vt[32];
#pragma unroll
      for (int i = 0; i < 8; ++i) {
        *(float4*)&kt[i * 4] = ((const float4*)kp)[i];
        *(float4*)&vt[i * 4] = ((const float4*)vp)[i];
      }
#pragma unroll
      for (int i = 0; i < 32; i += 2) {
        h2 val; val[0] = (_Float16)kt[i]; val[1] = (_Float16)kt[i + 1];
        *(h2*)&sK[row_s * KSTRIDE + d0 + i] = val;
      }
#pragma unroll
      for (int i = 0; i < 32; ++i)
        sV[(d0 + i) * VSTRIDE + row_s] = (_Float16)vt[i];
      if (s + 1 < T_ / 32) {   // prefetch next step's K/V lines into L2
        __builtin_prefetch(kp + 32 * H_ * D_, 0, 0);
        __builtin_prefetch(vp + 32 * H_ * D_, 0, 0);
      }
    }
    __syncthreads();

    // ---- S = Q * K^T : 2 M-tiles x 2 N-tiles, K-dim = 128 in 4 chunks.
    //      Each K B-operand load feeds two WMMAs (both M-tiles). ----
    v8f sc00 = 0.0f, sc01 = 0.0f, sc10 = 0.0f, sc11 = 0.0f;
#pragma unroll
    for (int c = 0; c < 4; ++c) {
      const int kb0 = (n)      * KSTRIDE + 32 * c + hi8;
      const int kb1 = (16 + n) * KSTRIDE + 32 * c + hi8;
      v8h a0 = *(const v8h*)&sK[kb0];
      v8h a1 = *(const v8h*)&sK[kb0 + 16];
      v8h b0 = *(const v8h*)&sK[kb1];
      v8h b1 = *(const v8h*)&sK[kb1 + 16];
      v16h kv0, kv1;
#pragma unroll
      for (int j = 0; j < 8; ++j) {
        kv0[j] = a0[j]; kv0[j + 8] = a1[j];
        kv1[j] = b0[j]; kv1[j + 8] = b1[j];
      }
      sc00 = wmma16(qa[0][c], kv0, sc00);
      sc01 = wmma16(qa[0][c], kv1, sc01);
      sc10 = wmma16(qa[1][c], kv0, sc10);
      sc11 = wmma16(qa[1][c], kv1, sc11);
    }

    // ---- alibi + dense bias + mask + online softmax, per M-tile ----
    const int col0 = key0 + n, col1 = key0 + 16 + n;
    const bool cp0 = col0 >= tlen, cp1 = col1 >= tlen;
    v16h pa[2];
#pragma unroll
    for (int mm = 0; mm < 2; ++mm) {
      v8f s0 = mm ? sc10 : sc00;
      v8f s1 = mm ? sc11 : sc01;
      float p0[8], p1[8], mx[8];
#pragma unroll
      for (int r = 0; r < 8; ++r) {
        const int row = m_base + mm * 16 + hi8 + r;
        const float* brow = bias + (size_t)b * T_ * T_ + (size_t)row * T_;
        float a0 = s0[r] - slope * fabsf((float)(row - col0)) + brow[col0];
        float a1 = s1[r] - slope * fabsf((float)(row - col1)) + brow[col1];
        const bool rp = row >= tlen;
        a0 = (rp || cp0) ? NEG_ : a0;
        a1 = (rp || cp1) ? NEG_ : a1;
        p0[r] = a0; p1[r] = a1;
        mx[r] = fmaxf(a0, a1);
      }
      // row-max across the 16-lane group (rows striped identically in both halves)
#pragma unroll
      for (int r = 0; r < 8; ++r) {
#pragma unroll
        for (int off = 1; off < 16; off <<= 1)
          mx[r] = fmaxf(mx[r], __shfl_xor(mx[r], off, 32));
      }
#pragma unroll
      for (int r = 0; r < 8; ++r) {
        const float mn = fmaxf(mrun[mm][r], mx[r]);
        const float scal = __expf(mrun[mm][r] - mn);
        mrun[mm][r] = mn;
        p0[r] = __expf(p0[r] - mn);
        p1[r] = __expf(p1[r] - mn);
        lrun[mm][r] *= scal;
#pragma unroll
        for (int d = 0; d < 8; ++d) acc[mm][d][r] *= scal;
      }

      // transpose P (C-layout -> A-layout) through per-wave LDS
      _Float16* pw = &sP[wave * 1024 + mm * 512];
#pragma unroll
      for (int r = 0; r < 8; ++r) {
        pw[(hi8 + r) * 32 + n]      = (_Float16)p0[r];
        pw[(hi8 + r) * 32 + 16 + n] = (_Float16)p1[r];
      }
      v8h lo = *(const v8h*)&pw[n * 32 + hi8];
      v8h hh = *(const v8h*)&pw[n * 32 + 16 + hi8];
#pragma unroll
      for (int j = 0; j < 8; ++j) { pa[mm][j] = lo[j]; pa[mm][j + 8] = hh[j]; }

      // row-sum(P) via WMMA against all-ones B: lands broadcast in C-layout
      v8f rsum = wmma16(pa[mm], ones, (v8f)0.0f);
#pragma unroll
      for (int r = 0; r < 8; ++r) lrun[mm][r] += rsum[r];
    }

    // ---- O += P * V : 8 N-tiles of 16; each V B-operand feeds both M-tiles ----
#pragma unroll
    for (int d = 0; d < 8; ++d) {
      const int vb = (d * 16 + n) * VSTRIDE + hi8;
      v8h x0 = *(const v8h*)&sV[vb];
      v8h x1 = *(const v8h*)&sV[vb + 16];
      v16h vv;
#pragma unroll
      for (int j = 0; j < 8; ++j) { vv[j] = x0[j]; vv[j + 8] = x1[j]; }
      acc[0][d] = wmma16(pa[0], vv, acc[0][d]);
      acc[1][d] = wmma16(pa[1], vv, acc[1][d]);
    }
  }

  // ---- normalize and write out (coalesced f32 rows) ----
#pragma unroll
  for (int mm = 0; mm < 2; ++mm)
#pragma unroll
    for (int r = 0; r < 8; ++r) {
      const int row = m_base + mm * 16 + hi8 + r;
      float* orow = out + (((size_t)b * T_ + row) * H_ + h) * D_;
      const float inv = 1.0f / lrun[mm][r];
#pragma unroll
      for (int d = 0; d < 8; ++d)
        orow[d * 16 + n] = acc[mm][d][r] * inv;
    }
}

extern "C" void kernel_launch(void* const* d_in, const int* in_sizes, int n_in,
                              void* d_out, int out_size, void* d_ws, size_t ws_size,
                              hipStream_t stream) {
  (void)in_sizes; (void)n_in; (void)out_size; (void)d_ws; (void)ws_size;
  const float* q      = (const float*)d_in[0];
  const float* k      = (const float*)d_in[1];
  const float* v      = (const float*)d_in[2];
  const int*   cum    = (const int*)d_in[3];
  const float* slopes = (const float*)d_in[4];
  const float* bias   = (const float*)d_in[5];
  float* out = (float*)d_out;

  dim3 grid(B_ * H_ * (T_ / 128));   // 512 workgroups
  dim3 block(128);                   // 4 waves of 32
  hipLaunchKernelGGL(fa_alibi_bias_kernel, grid, block, 0, stream,
                     q, k, v, cum, slopes, bias, out);
}
